// DecodeModel_33861522161885
// MI455X (gfx1250) — compile-verified, matrix operand order checked by reading, not run on hardware
//
#include <hip/hip_runtime.h>
#include <hip/hip_bf16.h>
#include <math.h>

// Model dims (compile-time constants from the reference)
#define LNUM 12
#define BB   8
#define HH   16
#define DD   64
#define EE   1024
#define FF   4096
#define VV   50257
#define SS   1024
#define TTC  1023
#define TALL 1024   // T + 1 (cached keys + new key)

typedef __attribute__((ext_vector_type(16))) __bf16 v16bf;
typedef __attribute__((ext_vector_type(16))) float  v16f;
typedef __attribute__((ext_vector_type(8)))  float  v8f;
typedef __attribute__((ext_vector_type(4)))  float  f4;

__device__ __forceinline__ float gelu_tanh(float x) {
    const float c = 0.7978845608028654f; // sqrt(2/pi)
    float x3 = x * x * x;
    return 0.5f * x * (1.0f + tanhf(c * (x + 0.044715f * x3)));
}

// ---------------------------------------------------------------------------
// WMMA GEMM: out[16 x N] = X[16 x K] @ W + bias (+gelu) (+resid)
// M is padded 8 -> 16 by CLAMPING row addresses (garbage rows/cols of D are
// simply never stored), so every load is unconditional -> deep MLP of loads
// in flight, no exec-mask churn.
// TRANSB=false: W row-major [K x N] (x @ W).   TRANSB=true: W [N x K] (x@W^T).
// ---------------------------------------------------------------------------
template <bool TRANSB, bool DOGELU, bool DORESID>
__global__ void wmma_gemm_kernel(const float* __restrict__ X,
                                 const float* __restrict__ W,
                                 const float* __restrict__ bias,
                                 const float* __restrict__ resid,
                                 float* __restrict__ out,
                                 int K, int N)
{
    int wave = blockIdx.x * (blockDim.x >> 5) + (threadIdx.x >> 5);
    int n0 = wave * 16;
    if (n0 >= N) return;                 // wave-uniform: EXEC stays full

    int lane = threadIdx.x & 31;
    int half = lane >> 4;                // 0: K-groups {0..7,16..23}; 1: {8..15,24..31}
    int lm   = lane & 15;
    int n    = n0 + lm;                  // this lane's B/C/D column
    bool nok = (n < N);
    int mc = (lm < BB) ? lm : (BB - 1);  // clamped A row (pad rows discarded later)
    int nc = nok ? n : (N - 1);          // clamped B column (pad cols discarded later)

    const float* xp = X + (size_t)mc * K + half * 8;
    const float* wp = TRANSB ? (W + (size_t)nc * K + half * 8)
                             : (W + (size_t)(half * 8) * N + nc);

    v8f acc = {};

#pragma unroll 2
    for (int k0 = 0; k0 < K; k0 += 32) {
        // ---- A fragment: 8+8 contiguous floats, two pairs of b128 loads
        f4 a0 = *(const f4*)(xp);
        f4 a1 = *(const f4*)(xp + 4);
        f4 a2 = *(const f4*)(xp + 16);
        f4 a3 = *(const f4*)(xp + 20);
        v16f af;
        af[0]=a0.x; af[1]=a0.y; af[2]=a0.z;  af[3]=a0.w;
        af[4]=a1.x; af[5]=a1.y; af[6]=a1.z;  af[7]=a1.w;
        af[8]=a2.x; af[9]=a2.y; af[10]=a2.z; af[11]=a2.w;
        af[12]=a3.x; af[13]=a3.y; af[14]=a3.z; af[15]=a3.w;

        // ---- B fragment (weights: streamed once -> non-temporal)
        v16f bf;
        if (TRANSB) {
            f4 b0 = __builtin_nontemporal_load((const f4*)(wp));
            f4 b1 = __builtin_nontemporal_load((const f4*)(wp + 4));
            f4 b2 = __builtin_nontemporal_load((const f4*)(wp + 16));
            f4 b3 = __builtin_nontemporal_load((const f4*)(wp + 20));
            bf[0]=b0.x; bf[1]=b0.y; bf[2]=b0.z;  bf[3]=b0.w;
            bf[4]=b1.x; bf[5]=b1.y; bf[6]=b1.z;  bf[7]=b1.w;
            bf[8]=b2.x; bf[9]=b2.y; bf[10]=b2.z; bf[11]=b2.w;
            bf[12]=b3.x; bf[13]=b3.y; bf[14]=b3.z; bf[15]=b3.w;
        } else {
#pragma unroll
            for (int i = 0; i < 8; ++i) {
                bf[i]     = __builtin_nontemporal_load(wp + (size_t)i * N);
                bf[i + 8] = __builtin_nontemporal_load(wp + (size_t)(16 + i) * N);
            }
        }

        v16bf a = __builtin_convertvector(af, v16bf);
        v16bf b = __builtin_convertvector(bf, v16bf);

        acc = __builtin_amdgcn_wmma_f32_16x16x32_bf16(
            /*neg_a=*/false, a, /*neg_b=*/false, b,
            /*c_mod=*/(short)0, acc, /*reuse_a=*/false, /*reuse_b=*/false);

        xp += 32;
        wp += TRANSB ? 32 : (size_t)32 * N;
    }

    if (!nok) return;                    // after all WMMAs; per-lane OK now
    float bv = bias ? bias[n] : 0.0f;
#pragma unroll
    for (int r = 0; r < 8; ++r) {
        int mr = r + half * 8;           // C/D layout: lanes>=16 hold rows 8..15
        if (mr >= BB) continue;
        float v = acc[r] + bv;
        if (DOGELU)  v = gelu_tanh(v);
        if (DORESID) v += resid[(size_t)mr * N + n];
        out[(size_t)mr * N + n] = v;
    }
}

// ---------------------------------------------------------------------------
// LayerNorm over E=1024, one block (256 threads) per row.
// ---------------------------------------------------------------------------
__global__ void ln_kernel(const float* __restrict__ in,
                          const float* __restrict__ w,
                          const float* __restrict__ b,
                          float* __restrict__ out)
{
    __shared__ float red[256];
    int row = blockIdx.x, tid = threadIdx.x;
    const float* x = in + (size_t)row * EE;

    float s = 0.0f;
    for (int e = tid; e < EE; e += 256) s += x[e];
    red[tid] = s; __syncthreads();
    for (int off = 128; off > 0; off >>= 1) {
        if (tid < off) red[tid] += red[tid + off];
        __syncthreads();
    }
    float mean = red[0] * (1.0f / EE);
    __syncthreads();

    float s2 = 0.0f;
    for (int e = tid; e < EE; e += 256) { float d = x[e] - mean; s2 += d * d; }
    red[tid] = s2; __syncthreads();
    for (int off = 128; off > 0; off >>= 1) {
        if (tid < off) red[tid] += red[tid + off];
        __syncthreads();
    }
    float rstd = rsqrtf(red[0] * (1.0f / EE) + 1e-5f);

    for (int e = tid; e < EE; e += 256)
        out[(size_t)row * EE + e] = (x[e] - mean) * rstd * w[e] + b[e];
}

// ---------------------------------------------------------------------------
// Embedding gather for the last token of each batch row.
// ---------------------------------------------------------------------------
__global__ void embed_kernel(const int* __restrict__ ids,
                             const float* __restrict__ wte,
                             const float* __restrict__ wpe,
                             float* __restrict__ h)
{
    int b = blockIdx.x;
    int tok = ids[(size_t)b * SS + (SS - 1)];
    for (int e = threadIdx.x; e < EE; e += blockDim.x)
        h[(size_t)b * EE + e] = wte[(size_t)tok * EE + e] + wpe[(size_t)(SS - 1) * EE + e];
}

// ---------------------------------------------------------------------------
// Decode attention, one block (256 thr = 8 waves) per (b,h).
// attn_bias is a constant shift along the softmax axis -> cancels, ignored.
// ---------------------------------------------------------------------------
__global__ void attn_kernel(const float* __restrict__ qkv,
                            const float* __restrict__ kc,
                            const float* __restrict__ vc,
                            float* __restrict__ ctx)
{
    __shared__ float s_q[DD];
    __shared__ float s_p[TALL];
    __shared__ float s_red[256];
    __shared__ float s_part[8 * DD];

    int bh = blockIdx.x;
    int b = bh / HH, h = bh % HH;
    int tid = threadIdx.x;

    const float* q    = qkv + (size_t)b * 3 * EE + h * DD;
    const float* knew = qkv + (size_t)b * 3 * EE + EE + h * DD;
    const float* vnew = qkv + (size_t)b * 3 * EE + 2 * EE + h * DD;
    const float* kbase = kc + ((size_t)(b * HH + h)) * TTC * DD;
    const float* vbase = vc + ((size_t)(b * HH + h)) * TTC * DD;

    if (tid < DD) s_q[tid] = q[tid];
    __syncthreads();

    // scores
    float lmax = -1e30f;
    for (int t = tid; t < TALL; t += 256) {
        const float* krow = (t < TTC) ? (kbase + (size_t)t * DD) : knew;
        const f4* k4 = (const f4*)krow;
        float dot = 0.0f;
#pragma unroll
        for (int d4 = 0; d4 < DD / 4; ++d4) {
            f4 kv = __builtin_nontemporal_load(k4 + d4);
            dot += s_q[4 * d4] * kv.x + s_q[4 * d4 + 1] * kv.y
                 + s_q[4 * d4 + 2] * kv.z + s_q[4 * d4 + 3] * kv.w;
        }
        dot *= 0.125f;                   // 1/sqrt(D=64)
        s_p[t] = dot;
        lmax = fmaxf(lmax, dot);
    }
    s_red[tid] = lmax; __syncthreads();
    for (int off = 128; off > 0; off >>= 1) {
        if (tid < off) s_red[tid] = fmaxf(s_red[tid], s_red[tid + off]);
        __syncthreads();
    }
    float gmax = s_red[0];
    __syncthreads();

    float lsum = 0.0f;
    for (int t = tid; t < TALL; t += 256) {
        float e = expf(s_p[t] - gmax);
        s_p[t] = e;
        lsum += e;
    }
    s_red[tid] = lsum; __syncthreads();
    for (int off = 128; off > 0; off >>= 1) {
        if (tid < off) s_red[tid] += s_red[tid + off];
        __syncthreads();
    }
    float inv = 1.0f / s_red[0];
    __syncthreads();

    // ctx = P @ V ; wave w handles 128 timesteps, lane l handles d = l, l+32
    int wv = tid >> 5, lane = tid & 31;
    float acc0 = 0.0f, acc1 = 0.0f;
    int t0 = wv * (TALL / 8);
    for (int t = t0; t < t0 + TALL / 8; ++t) {
        float p = s_p[t] * inv;
        const float* vrow = (t < TTC) ? (vbase + (size_t)t * DD) : vnew;
        acc0 += p * vrow[lane];
        acc1 += p * vrow[lane + 32];
    }
    s_part[wv * DD + lane]      = acc0;
    s_part[wv * DD + lane + 32] = acc1;
    __syncthreads();

    if (tid < DD) {
        float s = 0.0f;
#pragma unroll
        for (int w = 0; w < 8; ++w) s += s_part[w * DD + tid];
        ctx[(size_t)b * EE + h * DD + tid] = s;
    }
}

// ---------------------------------------------------------------------------
// Row softmax over V, in place (block per row).
// ---------------------------------------------------------------------------
__global__ void softmax_rows_kernel(float* __restrict__ x, int N)
{
    __shared__ float red[256];
    int row = blockIdx.x, tid = threadIdx.x;
    float* p = x + (size_t)row * N;

    float lmax = -1e30f;
    for (int i = tid; i < N; i += 256) lmax = fmaxf(lmax, p[i]);
    red[tid] = lmax; __syncthreads();
    for (int off = 128; off > 0; off >>= 1) {
        if (tid < off) red[tid] = fmaxf(red[tid], red[tid + off]);
        __syncthreads();
    }
    float gmax = red[0];
    __syncthreads();

    float lsum = 0.0f;
    for (int i = tid; i < N; i += 256) { float e = expf(p[i] - gmax); p[i] = e; lsum += e; }
    red[tid] = lsum; __syncthreads();
    for (int off = 128; off > 0; off >>= 1) {
        if (tid < off) red[tid] += red[tid + off];
        __syncthreads();
    }
    float inv = 1.0f / red[0];
    for (int i = tid; i < N; i += 256) p[i] *= inv;
}

// ---------------------------------------------------------------------------
extern "C" void kernel_launch(void* const* d_in, const int* in_sizes, int n_in,
                              void* d_out, int out_size, void* d_ws, size_t ws_size,
                              hipStream_t stream)
{
    const int*   ids  = (const int*)  d_in[0];
    const float* kc   = (const float*)d_in[1];
    const float* vc   = (const float*)d_in[2];
    const float* wte  = (const float*)d_in[3];
    const float* wpe  = (const float*)d_in[4];
    const float* ln1w = (const float*)d_in[5];
    const float* ln1b = (const float*)d_in[6];
    const float* caw  = (const float*)d_in[7];
    const float* cab  = (const float*)d_in[8];
    const float* apw  = (const float*)d_in[9];
    const float* apb  = (const float*)d_in[10];
    const float* ln2w = (const float*)d_in[11];
    const float* ln2b = (const float*)d_in[12];
    const float* fcw  = (const float*)d_in[13];
    const float* fcb  = (const float*)d_in[14];
    const float* pw   = (const float*)d_in[15];
    const float* pb   = (const float*)d_in[16];
    const float* lnfw = (const float*)d_in[17];
    const float* lnfb = (const float*)d_in[18];
    // d_in[19] attn_bias: constant along the softmax axis -> cancels; unused.

    float* ws   = (float*)d_ws;
    float* h    = ws;                 // 8*1024
    float* x    = h   + BB * EE;      // 8*1024
    float* qkv  = x   + BB * EE;      // 8*3072
    float* ctx  = qkv + BB * 3 * EE;  // 8*1024
    float* mbuf = ctx + BB * EE;      // 8*4096
    float* logits = (float*)d_out;    // 8*50257, softmaxed in place

    embed_kernel<<<BB, 256, 0, stream>>>(ids, wte, wpe, h);

    for (int l = 0; l < LNUM; ++l) {
        const float* kcl = kc + (size_t)l * BB * HH * TTC * DD;
        const float* vcl = vc + (size_t)l * BB * HH * TTC * DD;

        ln_kernel<<<BB, 256, 0, stream>>>(h, ln1w + (size_t)l * EE, ln1b + (size_t)l * EE, x);

        // qkv = x @ c_attn_w + c_attn_b      (K=1024, N=3072 -> 192 tiles)
        wmma_gemm_kernel<false, false, false><<<192 / 8, 256, 0, stream>>>(
            x, caw + (size_t)l * EE * 3 * EE, cab + (size_t)l * 3 * EE,
            nullptr, qkv, EE, 3 * EE);

        attn_kernel<<<BB * HH, 256, 0, stream>>>(qkv, kcl, vcl, ctx);

        // h += ctx @ attn_proj_w + attn_proj_b   (N=1024 -> 64 tiles)
        wmma_gemm_kernel<false, false, true><<<64 / 8, 256, 0, stream>>>(
            ctx, apw + (size_t)l * EE * EE, apb + (size_t)l * EE, h, h, EE, EE);

        ln_kernel<<<BB, 256, 0, stream>>>(h, ln2w + (size_t)l * EE, ln2b + (size_t)l * EE, x);

        // m = gelu(x @ fc_w + fc_b)          (K=1024, N=4096 -> 256 tiles)
        wmma_gemm_kernel<false, true, false><<<256 / 8, 256, 0, stream>>>(
            x, fcw + (size_t)l * EE * FF, fcb + (size_t)l * FF, nullptr, mbuf, EE, FF);

        // h += m @ mlp_proj_w + mlp_proj_b   (K=4096, N=1024 -> 64 tiles)
        wmma_gemm_kernel<false, false, true><<<64 / 8, 256, 0, stream>>>(
            mbuf, pw + (size_t)l * FF * EE, pb + (size_t)l * EE, h, h, FF, EE);
    }

    ln_kernel<<<BB, 256, 0, stream>>>(h, lnfw, lnfb, x);

    // logits = x @ wte^T   (TRANSB; K=1024, N=50257 -> 3142 tiles)
    {
        int ntile = (VV + 15) / 16;
        wmma_gemm_kernel<true, false, false><<<(ntile + 7) / 8, 256, 0, stream>>>(
            x, wte, nullptr, nullptr, logits, EE, VV);
    }

    softmax_rows_kernel<<<BB, 256, 0, stream>>>(logits, VV);
}
